// GreenFunctionResBlock_46978352283759
// MI455X (gfx1250) — matrix operand order repaired and checked by reading, not compiled, not run on hardware
//
#include <hip/hip_runtime.h>
#include <math.h>

// ---------------------------------------------------------------------------
// GreenFunctionResBlock fused kernels for MI455X (gfx1250, wave32, WMMA).
//   Kernel 1: attention MLP (6->32->32->1) over (N,P) with fused online
//             softmax + coeff accumulation; partials to workspace.
//   Kernel 2: merge softmax partials -> coeff; green MLP (10->64->64->64)
//             for g0 + 64 quadrature points; quad reduction; residual GEMM;
//             final 64->1 projection.
// All matrix math uses v_wmma_f32_16x16x32_f16 (f32 accumulate).
// All hot-path scalar math uses native v_exp_f32 / v_rcp_f32 / v_sqrt_f32
// (branch-free) so the loop stays WMMA/LDS-bound, not VALU-bound.
// ---------------------------------------------------------------------------

typedef __attribute__((ext_vector_type(16))) _Float16 v16h;
typedef __attribute__((ext_vector_type(8)))  _Float16 v8h;
typedef __attribute__((ext_vector_type(8)))  float    v8f;

#define NN 2048
#define PP 2048
#define QQ 64
#define PSLABS 4

#define LOG2E 1.4426950408889634f

__device__ __forceinline__ v8f wmma16(v16h a, v16h b, v8f c) {
  return __builtin_amdgcn_wmma_f32_16x16x32_f16(false, a, false, b,
                                                (short)0, c, false, false);
}

__device__ __forceinline__ v8f zero8() {
  v8f z;
#pragma unroll
  for (int i = 0; i < 8; ++i) z[i] = 0.0f;
  return z;
}

// exp(x) via single v_exp_f32. x*log2e underflows/overflows cleanly
// (exp2(-huge)=0, exp2(+huge)=inf), which is exactly what the online
// softmax (-1e30 masked logits, -inf initial max) needs.
__device__ __forceinline__ float fast_exp(float x) {
  return __builtin_amdgcn_exp2f(x * LOG2E);
}

// tanh(x) = 1 - 2/(exp(2x)+1), branch-free, saturates to +-1 for |x|->inf.
__device__ __forceinline__ float fast_tanh(float x) {
  const float e = __builtin_amdgcn_exp2f(x * (2.0f * LOG2E));
  return 1.0f - 2.0f * __builtin_amdgcn_rcpf(e + 1.0f);
}

// a/b via single v_rcp_f32 (b > 0 in all uses here).
__device__ __forceinline__ float fast_div(float a, float b) {
  return a * __builtin_amdgcn_rcpf(b);
}

__device__ __forceinline__ float fast_sqrt(float x) {
  return __builtin_amdgcn_sqrtf(x);
}

// Compiler + hardware fence for within-wave LDS relayout traffic.
__device__ __forceinline__ void lds_fence() {
  asm volatile("s_wait_dscnt 0" ::: "memory");
}

// Load one 16x32(K) f16 A-fragment from row-major LDS [16][ldh] halves.
// CDNA5 dense 16-bit A layout: lane holds row=(lane&15); element j maps to
// K = (j&7) + 8*hi + ((j&8)<<1)  => two contiguous 8-half (16B) chunks at
// half-offsets hi*8 and 16+hi*8 within the 32-wide K window.
__device__ __forceinline__ v16h load_afrag(const _Float16* base, int row,
                                           int hi, int ldh) {
  const v8h* p0 = (const v8h*)(base + row * ldh + hi * 8);
  const v8h* p1 = (const v8h*)(base + row * ldh + hi * 8 + 16);
  v8h lo = *p0;
  v8h h8 = *p1;
  return __builtin_shufflevector(lo, h8, 0, 1, 2, 3, 4, 5, 6, 7,
                                 8, 9, 10, 11, 12, 13, 14, 15);
}

// ---------------------------------------------------------------------------
// Kernel 1: attention + online softmax partials.
// Grid: (NN/16, PSLABS), block 256 (8 waves). Wave handles 16 n-rows, strided p.
// ---------------------------------------------------------------------------
__global__ __launch_bounds__(256) void attn_stage_kernel(
    const float* __restrict__ phase, const float* __restrict__ cpos,
    const float* __restrict__ cval, const float* __restrict__ aw1,
    const float* __restrict__ ab1, const float* __restrict__ aw2,
    const float* __restrict__ ab2, const float* __restrict__ aw3,
    const float* __restrict__ ab3,
    float* __restrict__ part /* [PSLABS][NN][4] */) {
  __shared__ __align__(16) _Float16 ldsA[8][16 * 32];
  __shared__ float logitbuf[8][16];
  __shared__ float mergebuf[8][16][4];

  const int lane = threadIdx.x & 31;
  const int wv = threadIdx.x >> 5;
  const int row = lane & 15;
  const int hi = lane >> 4;
  const int n = blockIdx.x * 16 + row;
  const int slab = blockIdx.y;

  const float x0 = phase[n * 4 + 0], x1 = phase[n * 4 + 1];
  const float vv0 = phase[n * 4 + 2], vv1 = phase[n * 4 + 3];
  const float vinv =
      __builtin_amdgcn_rcpf(fast_sqrt(vv0 * vv0 + vv1 * vv1 + 1e-16f));
  const float a0 = vv0 * vinv, a1 = vv1 * vinv;

  // Weight fragments (B layout: lane holds col=(lane&15); K = (j&7) + 8*hi + ((j&8)<<1)).
  v16h bf1[2], bf2[2], bf3;
#pragma unroll
  for (int t = 0; t < 2; ++t) {
    const int col = row + 16 * t;
#pragma unroll
    for (int j = 0; j < 16; ++j) {
      const int K = (j & 7) + hi * 8 + ((j & 8) << 1);
      bf1[t][j] = (K < 6) ? (_Float16)aw1[K * 32 + col] : (_Float16)0.0f;
      bf2[t][j] = (_Float16)aw2[K * 32 + col];
    }
  }
#pragma unroll
  for (int j = 0; j < 16; ++j) {
    const int K = (j & 7) + hi * 8 + ((j & 8) << 1);
    bf3[j] = (row == 0) ? (_Float16)aw3[K] : (_Float16)0.0f;
  }
  const float b1a = ab1[row], b1b = ab1[row + 16];
  const float b2a = ab2[row], b2b = ab2[row + 16];
  const float b3v = ab3[0];

  // Loop-invariant part of the layer-1 A fragment (K=0..3 of hi==0 lanes).
  v16h afbase;
#pragma unroll
  for (int j = 0; j < 16; ++j) afbase[j] = (_Float16)0.0f;
  if (hi == 0) {
    afbase[0] = (_Float16)x0;
    afbase[1] = (_Float16)x1;
    afbase[2] = (_Float16)vv0;
    afbase[3] = (_Float16)vv1;
  }

  float mrun = -INFINITY, ssum = 0.0f, ac0 = 0.0f, ac1 = 0.0f;

  _Float16* myA = ldsA[wv];
  float* mylg = logitbuf[wv];
  const v8f z = zero8();

  for (int it = 0; it < (PP / PSLABS) / 8; ++it) {
    const int p = slab * (PP / PSLABS) + wv + 8 * it;
    const float cp0 = cpos[p * 2 + 0], cp1 = cpos[p * 2 + 1];
    const float cv0 = cval[p * 2 + 0], cv1 = cval[p * 2 + 1];
    const float r0 = x0 - cp0, r1 = x1 - cp1;
    const float dist = fast_sqrt(r0 * r0 + r1 * r1 + 1e-16f);
    const float pos_loc = r0 * a0 + r1 * a1;
    const float ang_loc = fast_div(pos_loc, dist + 1e-8f);

    // A fragment, layer 1 (K padded 6->32; only hi==0 lanes carry K=0..5).
    v16h af = afbase;
    if (hi == 0) {
      af[4] = (_Float16)ang_loc;
      af[5] = (_Float16)pos_loc;
    }
    v8f d0 = wmma16(af, bf1[0], z);
    v8f d1 = wmma16(af, bf1[1], z);

    lds_fence();  // WAR vs previous iteration's fragment reads
#pragma unroll
    for (int r = 0; r < 8; ++r) {
      const int m = r + 8 * hi;
      myA[m * 32 + row] = (_Float16)fast_tanh(d0[r] + b1a);
      myA[m * 32 + row + 16] = (_Float16)fast_tanh(d1[r] + b1b);
    }
    lds_fence();
    v16h a2 = load_afrag(myA, row, hi, 32);
    v8f e0 = wmma16(a2, bf2[0], z);
    v8f e1 = wmma16(a2, bf2[1], z);
    lds_fence();
#pragma unroll
    for (int r = 0; r < 8; ++r) {
      const int m = r + 8 * hi;
      myA[m * 32 + row] = (_Float16)fast_tanh(e0[r] + b2a);
      myA[m * 32 + row + 16] = (_Float16)fast_tanh(e1[r] + b2b);
    }
    lds_fence();
    v16h a3 = load_afrag(myA, row, hi, 32);
    v8f d3 = wmma16(a3, bf3, z);
    lds_fence();
    if (row == 0) {  // column 0 lives in lanes 0 (rows 0-7) and 16 (rows 8-15)
#pragma unroll
      for (int r = 0; r < 8; ++r) mylg[r + 8 * hi] = d3[r] + b3v;
    }
    lds_fence();
    float lg = mylg[row];
    lg = (pos_loc > 0.0f) ? lg : -1e30f;

    // Online softmax + coeff-value accumulation (all per-lane, branch-free).
    const float mn = fmaxf(mrun, lg);
    const float cc = fast_exp(mrun - mn);
    const float ee = fast_exp(lg - mn);
    ssum = ssum * cc + ee;
    ac0 = ac0 * cc + ee * cv0;
    ac1 = ac1 * cc + ee * cv1;
    mrun = mn;
  }

  if (lane < 16) {
    mergebuf[wv][row][0] = mrun;
    mergebuf[wv][row][1] = ssum;
    mergebuf[wv][row][2] = ac0;
    mergebuf[wv][row][3] = ac1;
  }
  __syncthreads();
  if (wv == 0 && lane < 16) {
    float M = -INFINITY, S = 0.0f, A0 = 0.0f, A1 = 0.0f;
    for (int w = 0; w < 8; ++w) {
      const float mw = mergebuf[w][row][0];
      const float sw = mergebuf[w][row][1];
      const float p0 = mergebuf[w][row][2];
      const float p1 = mergebuf[w][row][3];
      const float Mn = fmaxf(M, mw);
      const float c1 = fast_exp(M - Mn);
      const float c2 = fast_exp(mw - Mn);
      S = S * c1 + sw * c2;
      A0 = A0 * c1 + p0 * c2;
      A1 = A1 * c1 + p1 * c2;
      M = Mn;
    }
    float* dst = part + ((size_t)slab * NN + n) * 4;
    dst[0] = M;
    dst[1] = S;
    dst[2] = A0;
    dst[3] = A1;
  }
}

// ---------------------------------------------------------------------------
// Kernel 2: coeff merge + green MLP + quad + residual + final projection.
// Grid: NN/16, block 256 (8 waves). Block owns 16 n-rows.
// ---------------------------------------------------------------------------
__global__ __launch_bounds__(256) void green_stage_kernel(
    const float* __restrict__ phase, const float* __restrict__ bound,
    const float* __restrict__ skc, const float* __restrict__ sk,
    const float* __restrict__ vwq, const float* __restrict__ gw1,
    const float* __restrict__ gb1, const float* __restrict__ gw2,
    const float* __restrict__ gb2, const float* __restrict__ gw3,
    const float* __restrict__ gb3, const float* __restrict__ rw,
    const float* __restrict__ rb, const float* __restrict__ fw,
    const float* __restrict__ fb, const float* __restrict__ part,
    float* __restrict__ out) {
  __shared__ __align__(16) _Float16 ldsA[8][16 * 64];
  __shared__ float quadbuf[8][16][64];
  __shared__ float g0buf[16][64];

  const int lane = threadIdx.x & 31;
  const int wv = threadIdx.x >> 5;
  const int row = lane & 15;
  const int hi = lane >> 4;
  const int n = blockIdx.x * 16 + row;

  // Merge softmax partials -> coeff (per lane, duplicated across halves).
  float M = -INFINITY, S = 0.0f, A0 = 0.0f, A1 = 0.0f;
#pragma unroll
  for (int s = 0; s < PSLABS; ++s) {
    const float* src = part + ((size_t)s * NN + n) * 4;
    const float mw = src[0], sw = src[1], p0 = src[2], p1 = src[3];
    const float Mn = fmaxf(M, mw);
    const float c1 = fast_exp(M - Mn);
    const float c2 = fast_exp(mw - Mn);
    S = S * c1 + sw * c2;
    A0 = A0 * c1 + p0 * c2;
    A1 = A1 * c1 + p1 * c2;
    M = Mn;
  }
  const float sinv = __builtin_amdgcn_rcpf(S);
  const float coeff0 = fast_exp(-A0 * sinv);
  const float coeff1 = fast_exp(-A1 * sinv);

  const float x0 = phase[n * 4 + 0], x1 = phase[n * 4 + 1];
  const float v0 = phase[n * 4 + 2], v1 = phase[n * 4 + 3];
  const float xp0 = bound[n * 4 + 0], xp1 = bound[n * 4 + 1];
  const float vp0 = bound[n * 4 + 2], vp1 = bound[n * 4 + 3];

  // Green weight fragments. Layer1: 10->64 (K padded to 32), 4 N-tiles.
  // Layers 2/3: 64->64, chained K=32 pairs x 4 N-tiles.
  v16h bg1[4], bg2[2][4], bg3[2][4];
  float bias1[4], bias2[4], bias3[4];
#pragma unroll
  for (int t = 0; t < 4; ++t) {
    const int col = row + 16 * t;
#pragma unroll
    for (int j = 0; j < 16; ++j) {
      const int K = (j & 7) + 8 * hi + ((j & 8) << 1);
      bg1[t][j] = (K < 10) ? (_Float16)gw1[K * 64 + col] : (_Float16)0.0f;
      bg2[0][t][j] = (_Float16)gw2[K * 64 + col];
      bg2[1][t][j] = (_Float16)gw2[(K + 32) * 64 + col];
      bg3[0][t][j] = (_Float16)gw3[K * 64 + col];
      bg3[1][t][j] = (_Float16)gw3[(K + 32) * 64 + col];
    }
    bias1[t] = gb1[col];
    bias2[t] = gb2[col];
    bias3[t] = gb3[col];
  }

  v8f qacc[4];
#pragma unroll
  for (int t = 0; t < 4; ++t) qacc[t] = zero8();

  _Float16* myA = ldsA[wv];
  const v8f z = zero8();

  // Variant 0 = g0 input; variants 1..64 = quadrature points.
  for (int v = wv; v <= QQ; v += 8) {
    float f6, f7, wq = 0.0f;
    if (v == 0) {
      f6 = vp0;
      f7 = vp1;
    } else {
      const int q = v - 1;
      f6 = skc[q * 2 + 0];
      f7 = skc[q * 2 + 1];
      wq = (1.0f - sk[q]) * vwq[q];
    }
    v16h af;
#pragma unroll
    for (int j = 0; j < 16; ++j) af[j] = (_Float16)0.0f;
    if (hi == 0) {  // K=0..7
      af[0] = (_Float16)x0;
      af[1] = (_Float16)x1;
      af[2] = (_Float16)v0;
      af[3] = (_Float16)v1;
      af[4] = (_Float16)xp0;
      af[5] = (_Float16)xp1;
      af[6] = (_Float16)f6;
      af[7] = (_Float16)f7;
    } else {  // K=8..15 -> feat[8], feat[9]
      af[0] = (_Float16)coeff0;
      af[1] = (_Float16)coeff1;
    }
    v8f d[4];
#pragma unroll
    for (int t = 0; t < 4; ++t) d[t] = wmma16(af, bg1[t], z);
    lds_fence();
#pragma unroll
    for (int t = 0; t < 4; ++t)
#pragma unroll
      for (int r = 0; r < 8; ++r)
        myA[(r + 8 * hi) * 64 + row + 16 * t] =
            (_Float16)fast_tanh(d[t][r] + bias1[t]);
    lds_fence();
    v16h ak0 = load_afrag(myA, row, hi, 64);
    v16h ak1 = load_afrag(myA + 32, row, hi, 64);
#pragma unroll
    for (int t = 0; t < 4; ++t) {
      d[t] = wmma16(ak0, bg2[0][t], z);
      d[t] = wmma16(ak1, bg2[1][t], d[t]);
    }
    lds_fence();
#pragma unroll
    for (int t = 0; t < 4; ++t)
#pragma unroll
      for (int r = 0; r < 8; ++r)
        myA[(r + 8 * hi) * 64 + row + 16 * t] =
            (_Float16)fast_tanh(d[t][r] + bias2[t]);
    lds_fence();
    ak0 = load_afrag(myA, row, hi, 64);
    ak1 = load_afrag(myA + 32, row, hi, 64);
#pragma unroll
    for (int t = 0; t < 4; ++t) {
      d[t] = wmma16(ak0, bg3[0][t], z);
      d[t] = wmma16(ak1, bg3[1][t], d[t]);
    }
    if (v == 0) {  // only wave 0 ever takes this branch
#pragma unroll
      for (int t = 0; t < 4; ++t)
#pragma unroll
        for (int r = 0; r < 8; ++r)
          g0buf[r + 8 * hi][row + 16 * t] =
              fast_exp(fast_tanh(d[t][r] + bias3[t]));
    } else {
#pragma unroll
      for (int t = 0; t < 4; ++t)
#pragma unroll
        for (int r = 0; r < 8; ++r)
          qacc[t][r] += wq * fast_exp(fast_tanh(d[t][r] + bias3[t]));
    }
  }

#pragma unroll
  for (int t = 0; t < 4; ++t)
#pragma unroll
    for (int r = 0; r < 8; ++r)
      quadbuf[wv][r + 8 * hi][row + 16 * t] = qacc[t][r];
  __syncthreads();

  if (wv == 0) {
    // Reduce quad across waves straight into D-layout registers.
    v8f qs[4];
#pragma unroll
    for (int t = 0; t < 4; ++t)
#pragma unroll
      for (int r = 0; r < 8; ++r) {
        float s = 0.0f;
        for (int w = 0; w < 8; ++w) s += quadbuf[w][r + 8 * hi][row + 16 * t];
        qs[t][r] = s;
      }
    lds_fence();
#pragma unroll
    for (int t = 0; t < 4; ++t)
#pragma unroll
      for (int r = 0; r < 8; ++r)
        myA[(r + 8 * hi) * 64 + row + 16 * t] = (_Float16)qs[t][r];
    lds_fence();
    v16h ak0 = load_afrag(myA, row, hi, 64);
    v16h ak1 = load_afrag(myA + 32, row, hi, 64);

    v16h br[2][4];
    float rbias[4];
#pragma unroll
    for (int t = 0; t < 4; ++t) {
      const int col = row + 16 * t;
#pragma unroll
      for (int j = 0; j < 16; ++j) {
        const int K = (j & 7) + 8 * hi + ((j & 8) << 1);
        br[0][t][j] = (_Float16)rw[K * 64 + col];
        br[1][t][j] = (_Float16)rw[(K + 32) * 64 + col];
      }
      rbias[t] = rb[col];
    }
    v8f dr[4];
#pragma unroll
    for (int t = 0; t < 4; ++t) {
      dr[t] = wmma16(ak0, br[0][t], z);
      dr[t] = wmma16(ak1, br[1][t], dr[t]);
    }
    lds_fence();
#pragma unroll
    for (int t = 0; t < 4; ++t)
#pragma unroll
      for (int r = 0; r < 8; ++r) {
        const float val =
            fast_tanh(dr[t][r] + rbias[t]) + g0buf[r + 8 * hi][row + 16 * t];
        myA[(r + 8 * hi) * 64 + row + 16 * t] = (_Float16)val;
      }
    lds_fence();
    ak0 = load_afrag(myA, row, hi, 64);
    ak1 = load_afrag(myA + 32, row, hi, 64);
    v16h bfin[2];
#pragma unroll
    for (int j = 0; j < 16; ++j) {
      const int K = (j & 7) + 8 * hi + ((j & 8) << 1);
      bfin[0][j] = (row == 0) ? (_Float16)fw[K] : (_Float16)0.0f;
      bfin[1][j] = (row == 0) ? (_Float16)fw[K + 32] : (_Float16)0.0f;
    }
    v8f df = wmma16(ak0, bfin[0], z);
    df = wmma16(ak1, bfin[1], df);
    if (row == 0) {
      const float fbv = fb[0];
#pragma unroll
      for (int r = 0; r < 8; ++r)
        out[blockIdx.x * 16 + r + 8 * hi] = df[r] + fbv;
    }
  }
}

// ---------------------------------------------------------------------------
extern "C" void kernel_launch(void* const* d_in, const int* in_sizes, int n_in,
                              void* d_out, int out_size, void* d_ws,
                              size_t ws_size, hipStream_t stream) {
  (void)in_sizes;
  (void)n_in;
  (void)out_size;
  (void)ws_size;
  const float* phase = (const float*)d_in[0];
  const float* bound = (const float*)d_in[1];
  const float* cpos = (const float*)d_in[2];
  const float* cval = (const float*)d_in[3];
  const float* skc = (const float*)d_in[4];
  const float* sk = (const float*)d_in[5];
  const float* vwq = (const float*)d_in[6];
  const float* aw1 = (const float*)d_in[7];
  const float* ab1 = (const float*)d_in[8];
  const float* aw2 = (const float*)d_in[9];
  const float* ab2 = (const float*)d_in[10];
  const float* aw3 = (const float*)d_in[11];
  const float* ab3 = (const float*)d_in[12];
  const float* gw1 = (const float*)d_in[13];
  const float* gb1 = (const float*)d_in[14];
  const float* gw2 = (const float*)d_in[15];
  const float* gb2 = (const float*)d_in[16];
  const float* gw3 = (const float*)d_in[17];
  const float* gb3 = (const float*)d_in[18];
  const float* rw = (const float*)d_in[19];
  const float* rb = (const float*)d_in[20];
  const float* fw = (const float*)d_in[21];
  const float* fb = (const float*)d_in[22];

  float* part = (float*)d_ws;  // PSLABS * NN * 4 floats = 128 KB
  float* out = (float*)d_out;  // NN floats

  dim3 grid1(NN / 16, PSLABS);
  attn_stage_kernel<<<grid1, 256, 0, stream>>>(phase, cpos, cval, aw1, ab1,
                                               aw2, ab2, aw3, ab3, part);
  green_stage_kernel<<<NN / 16, 256, 0, stream>>>(
      phase, bound, skc, sk, vwq, gw1, gb1, gw2, gb2, gw3, gb3, rw, rb, fw, fb,
      part, out);
}
